// BevFormer_62156766708014
// MI455X (gfx1250) — matrix operand order repaired and checked by reading, not compiled, not run on hardware
//
#include <hip/hip_runtime.h>
#include <hip/hip_bf16.h>

// ---------------- problem constants ----------------
#define B_   2
#define Q_   2500
#define P_   2500
#define KL_  2500
#define E_   256
#define H_   8
#define D_   32
#define FF_  512
#define M_   (B_ * Q_)          // 5000 query rows
#define NPAD_ 2528              // kv length padded to multiple of 32

typedef __bf16 bf16_t;
typedef __attribute__((ext_vector_type(16))) __bf16 v16bf;
typedef __attribute__((ext_vector_type(8)))  __bf16 v8bf;
typedef __attribute__((ext_vector_type(8)))  float  v8f;

union vf8u  { v8f  v; float  f[8];  };
union vb16u { v16bf v; v8bf h[2]; };

// A-fragment (16x32 bf16): lane row = rowptr; two 8-elem chunks.
// lanes 0-15: K {0..7, 16..23}; lanes 16-31: K {8..15, 24..31}.
static __device__ __forceinline__ v16bf load_a_frag(const bf16_t* rowptr, int lane) {
  const int kb = (lane & 16) ? 8 : 0;
  vb16u u;
  u.h[0] = *(const v8bf*)(rowptr + kb);
  u.h[1] = *(const v8bf*)(rowptr + kb + 16);
  return u.v;
}
// B-fragment (32x16 bf16): lane column = colptr (length-32 K run);
// lanes 0-15: K {0..7, 16..23}; lanes 16-31: K {8..15, 24..31}.
static __device__ __forceinline__ v16bf load_b_frag(const bf16_t* colptr, int lane) {
  const int kb = (lane & 16) ? 8 : 0;
  vb16u u;
  u.h[0] = *(const v8bf*)(colptr + kb);
  u.h[1] = *(const v8bf*)(colptr + kb + 16);
  return u.v;
}

static __device__ __forceinline__ v8f wmma_bf16(v16bf a, v16bf b, v8f c) {
  return __builtin_amdgcn_wmma_f32_16x16x32_bf16(false, a, false, b, (short)0, c,
                                                 false, false);
}

// ---------------- elementwise helpers ----------------
__global__ void k_f32_to_bf16(const float* __restrict__ x, bf16_t* __restrict__ y, int n) {
  int i = blockIdx.x * blockDim.x + threadIdx.x;
  if (i < n) y[i] = (bf16_t)x[i];
}

// W[K][N] f32 -> WT[N][K] bf16
__global__ void k_wtrans(const float* __restrict__ W, bf16_t* __restrict__ WT, int K, int N) {
  int i = blockIdx.x * blockDim.x + threadIdx.x;
  if (i >= N * K) return;
  int n = i / K, k = i - n * K;
  WT[i] = (bf16_t)W[(size_t)k * N + n];
}

// Vp (b, kv, h, d) bf16  ->  VT (b, h, d, kvpad) bf16, zero padded
__global__ void k_vtrans(const bf16_t* __restrict__ Vp, bf16_t* __restrict__ VT, int klen) {
  int i = blockIdx.x * blockDim.x + threadIdx.x;
  int total = B_ * H_ * D_ * NPAD_;
  if (i >= total) return;
  int kv = i % NPAD_; int t = i / NPAD_;
  int d = t % D_; t /= D_;
  int h = t % H_; int b = t / H_;
  bf16_t v = (bf16_t)0.0f;
  if (kv < klen) v = Vp[((size_t)(b * klen + kv) * H_ + h) * D_ + d];
  VT[i] = v;
}

// ---------------- generic bf16 WMMA GEMM ----------------
// C[M,N] = A[M,K](row-major bf16) x BT[N,K]^T (+bias)(+relu).
// Wave computes a 16x64 tile: one A fragment feeds 4 WMMAs (4x A-traffic reuse).
__global__ void k_gemm(const bf16_t* __restrict__ A, const bf16_t* __restrict__ BT,
                       const float* __restrict__ bias, float* __restrict__ Cf,
                       bf16_t* __restrict__ Cb, int M, int N, int K, int relu) {
  const int lane = threadIdx.x & 31;
  const int wid  = (blockIdx.x * blockDim.x + threadIdx.x) >> 5;
  const int tiles_n = N >> 6;                  // 64-column tiles
  const int tm = wid / tiles_n, tn = wid - tm * tiles_n;
  const int m0 = tm << 4, n0 = tn << 6;
  if (m0 >= M) return;                         // whole-wave exit: EXEC stays all-ones
  int arow = m0 + (lane & 15); if (arow >= M) arow = M - 1;
  const bf16_t* ap  = A  + (size_t)arow * K;
  const bf16_t* bp0 = BT + (size_t)(n0 +  0 + (lane & 15)) * K;
  const bf16_t* bp1 = BT + (size_t)(n0 + 16 + (lane & 15)) * K;
  const bf16_t* bp2 = BT + (size_t)(n0 + 32 + (lane & 15)) * K;
  const bf16_t* bp3 = BT + (size_t)(n0 + 48 + (lane & 15)) * K;
  v8f a0 = {}, a1 = {}, a2 = {}, a3 = {};
  for (int kk = 0; kk < K; kk += 32) {
    v16bf af = load_a_frag(ap + kk, lane);
    a0 = wmma_bf16(af, load_b_frag(bp0 + kk, lane), a0);
    a1 = wmma_bf16(af, load_b_frag(bp1 + kk, lane), a1);
    a2 = wmma_bf16(af, load_b_frag(bp2 + kk, lane), a2);
    a3 = wmma_bf16(af, load_b_frag(bp3 + kk, lane), a3);
  }
  vf8u c[4]; c[0].v = a0; c[1].v = a1; c[2].v = a2; c[3].v = a3;
  const int mhalf = (lane >> 4) << 3;
#pragma unroll
  for (int j = 0; j < 4; ++j) {
    const int nidx = n0 + (j << 4) + (lane & 15);
    const float bv = bias ? bias[nidx] : 0.0f;
#pragma unroll
    for (int v = 0; v < 8; ++v) {
      int m = m0 + v + mhalf;
      if (m >= M) continue;
      float y = c[j].f[v] + bv;
      if (relu) y = fmaxf(y, 0.0f);
      size_t idx = (size_t)m * N + nidx;
      if (Cf) Cf[idx] = y;
      if (Cb) Cb[idx] = (bf16_t)y;
    }
  }
}

// ---------------- flash attention helpers ----------------
static __device__ __forceinline__ void softmax_update(
    const vf8u& s0, const vf8u& s1, float sc, bool o0, bool o1,
    float* m, float* l, vf8u& acc0, vf8u& acc1, bf16_t* lp, int lane) {
  const int r0 = (lane >> 4) << 3;
#pragma unroll
  for (int v = 0; v < 8; ++v) {
    float a0 = o0 ? -1e30f : s0.f[v] * sc;
    float a1 = o1 ? -1e30f : s1.f[v] * sc;
    float tmax = fmaxf(a0, a1);
    for (int off = 1; off < 16; off <<= 1)
      tmax = fmaxf(tmax, __shfl_xor(tmax, off, 16));
    float mn    = fmaxf(m[v], tmax);
    float alpha = __expf(m[v] - mn);
    float p0 = __expf(a0 - mn);
    float p1 = __expf(a1 - mn);
    float rs = p0 + p1;
    for (int off = 1; off < 16; off <<= 1)
      rs += __shfl_xor(rs, off, 16);
    l[v] = l[v] * alpha + rs;
    m[v] = mn;
    acc0.f[v] *= alpha;
    acc1.f[v] *= alpha;
    int r = v + r0;                            // C-layout row for this vgpr
    lp[r * 32 + (lane & 15)]      = (bf16_t)p0;
    lp[r * 32 + 16 + (lane & 15)] = (bf16_t)p1;
  }
}

static __device__ __forceinline__ void store_o(
    float* __restrict__ Of, bf16_t* __restrict__ Ob,
    const vf8u& acc0, const vf8u& acc1, const float* l,
    int b, int h, int qbase, int qlen, int lane) {
  const int mhalf = (lane >> 4) << 3;
#pragma unroll
  for (int v = 0; v < 8; ++v) {
    int q = qbase + v + mhalf;
    if (q >= qlen) continue;
    float inv = 1.0f / l[v];
    size_t base = ((size_t)(b * qlen + q) * H_ + h) * D_;
    float o0 = acc0.f[v] * inv, o1 = acc1.f[v] * inv;
    if (Of) { Of[base + (lane & 15)] = o0; Of[base + 16 + (lane & 15)] = o1; }
    if (Ob) { Ob[base + (lane & 15)] = (bf16_t)o0; Ob[base + 16 + (lane & 15)] = (bf16_t)o1; }
  }
}

// ---------------- flash attention (wave = 32-row q tile) ----------------
// Qp,Kp: (b, seq, h, d) bf16; VT: (b, h, d, NPAD_) bf16.
// Each K/V fragment feeds 2 WMMAs (two q sub-tiles) -> halves K/V L2 traffic.
__global__ void k_flash(const bf16_t* __restrict__ Qp, const bf16_t* __restrict__ Kp,
                        const bf16_t* __restrict__ VT, float* __restrict__ Of,
                        bf16_t* __restrict__ Ob, int qlen, int klen, float sc) {
  __shared__ alignas(64) bf16_t plds[8][2 * 512];
  const int lane = threadIdx.x & 31;
  const int wib  = threadIdx.x >> 5;
  const int wid  = blockIdx.x * 8 + wib;
  const int tiles_q = (qlen + 31) >> 5;
  const int bh = wid / tiles_q;
  if (bh >= B_ * H_) return;                   // whole-wave exit
  const int b = bh >> 3, h = bh & 7;
  const int q0 = (wid - bh * tiles_q) << 5;

  int qa = q0 + (lane & 15);      if (qa >= qlen) qa = qlen - 1;
  int qb = q0 + 16 + (lane & 15); if (qb >= qlen) qb = qlen - 1;
  const v16bf qfA = load_a_frag(Qp + ((size_t)(b * qlen + qa) * H_ + h) * D_, lane);
  const v16bf qfB = load_a_frag(Qp + ((size_t)(b * qlen + qb) * H_ + h) * D_, lane);

  float mA[8], lA[8], mB[8], lB[8];
  vf8u acc0A, acc1A, acc0B, acc1B;
  acc0A.v = (v8f){}; acc1A.v = (v8f){}; acc0B.v = (v8f){}; acc1B.v = (v8f){};
#pragma unroll
  for (int v = 0; v < 8; ++v) { mA[v] = -1e30f; lA[v] = 0.0f; mB[v] = -1e30f; lB[v] = 0.0f; }

  bf16_t* lpA = plds[wib];
  bf16_t* lpB = plds[wib] + 512;
  const bf16_t* vbase = VT + ((size_t)(b * H_ + h) * D_) * NPAD_;

  for (int kv0 = 0; kv0 < klen; kv0 += 32) {
    // ---- shared K fragments, 4 score WMMAs ----
    int c0 = kv0 + (lane & 15); int cc0 = c0 < klen ? c0 : klen - 1;
    int c1 = c0 + 16;           int cc1 = c1 < klen ? c1 : klen - 1;
    v16bf kf0 = load_b_frag(Kp + ((size_t)(b * klen + cc0) * H_ + h) * D_, lane);
    v16bf kf1 = load_b_frag(Kp + ((size_t)(b * klen + cc1) * H_ + h) * D_, lane);
    vf8u s0A, s1A, s0B, s1B;
    s0A.v = wmma_bf16(qfA, kf0, (v8f){});
    s1A.v = wmma_bf16(qfA, kf1, (v8f){});
    s0B.v = wmma_bf16(qfB, kf0, (v8f){});
    s1B.v = wmma_bf16(qfB, kf1, (v8f){});
    const bool o0 = c0 >= klen, o1 = c1 >= klen;
    // ---- streaming softmax per 16-row sub-tile, P -> LDS (bf16) ----
    softmax_update(s0A, s1A, sc, o0, o1, mA, lA, acc0A, acc1A, lpA, lane);
    softmax_update(s0B, s1B, sc, o0, o1, mB, lB, acc0B, acc1B, lpB, lane);
    asm volatile("s_wait_dscnt 0x0" ::: "memory");   // cross-lane LDS visibility
    // ---- re-layout P (C-frag -> A-frag), shared V fragments, 4 PV WMMAs ----
    v16bf pfA = load_a_frag(lpA + (size_t)(lane & 15) * 32, lane);
    v16bf pfB = load_a_frag(lpB + (size_t)(lane & 15) * 32, lane);
    v16bf vf0 = load_b_frag(vbase + (size_t)(lane & 15) * NPAD_ + kv0, lane);
    v16bf vf1 = load_b_frag(vbase + (size_t)(16 + (lane & 15)) * NPAD_ + kv0, lane);
    acc0A.v = wmma_bf16(pfA, vf0, acc0A.v);
    acc1A.v = wmma_bf16(pfA, vf1, acc1A.v);
    acc0B.v = wmma_bf16(pfB, vf0, acc0B.v);
    acc1B.v = wmma_bf16(pfB, vf1, acc1B.v);
  }
  store_o(Of, Ob, acc0A, acc1A, lA, b, h, q0,      qlen, lane);
  store_o(Of, Ob, acc0B, acc1B, lB, b, h, q0 + 16, qlen, lane);
}

// ---------------- residual add + layernorm (wave per 256-wide row) ----------------
__global__ void k_add_ln(const float* __restrict__ X, const float* __restrict__ R,
                         const float* __restrict__ g, const float* __restrict__ bta,
                         float* __restrict__ Yf, bf16_t* __restrict__ Yb, int rows) {
  const int lane = threadIdx.x & 31;
  const int row  = blockIdx.x * 8 + (threadIdx.x >> 5);
  if (row >= rows) return;
  const float* xp = X + (size_t)row * E_;
  const float* rp = R + (size_t)row * E_;
  float vals[8], s = 0.0f, ss = 0.0f;
#pragma unroll
  for (int j = 0; j < 8; ++j) {
    float v = xp[j * 32 + lane] + rp[j * 32 + lane];
    vals[j] = v; s += v; ss += v * v;
  }
  for (int off = 1; off < 32; off <<= 1) {
    s  += __shfl_xor(s,  off, 32);
    ss += __shfl_xor(ss, off, 32);
  }
  const float mean = s * (1.0f / E_);
  const float var  = ss * (1.0f / E_) - mean * mean;
  const float rstd = rsqrtf(var + 1e-5f);
#pragma unroll
  for (int j = 0; j < 8; ++j) {
    int i = j * 32 + lane;
    float y = (vals[j] - mean) * rstd * g[i] + bta[i];
    if (Yf) Yf[(size_t)row * E_ + i] = y;
    if (Yb) Yb[(size_t)row * E_ + i] = (bf16_t)y;
  }
}

// ---------------- host orchestration ----------------
static inline int cdiv_i(long a, long b) { return (int)((a + b - 1) / b); }

extern "C" void kernel_launch(void* const* d_in, const int* in_sizes, int n_in,
                              void* d_out, int out_size, void* d_ws, size_t ws_size,
                              hipStream_t stream) {
  (void)in_sizes; (void)n_in; (void)out_size; (void)ws_size;
  const float* prev  = (const float*)d_in[0];
  const float* query = (const float*)d_in[1];
  const float* img   = (const float*)d_in[2];
  const float* t_wq  = (const float*)d_in[3];
  const float* t_wk  = (const float*)d_in[4];
  const float* t_wv  = (const float*)d_in[5];
  const float* s_wq  = (const float*)d_in[6];
  const float* s_wk  = (const float*)d_in[7];
  const float* s_wv  = (const float*)d_in[8];
  const float* s_wo  = (const float*)d_in[9];
  const float* s_bo  = (const float*)d_in[10];
  const float* n1_g  = (const float*)d_in[11];
  const float* n1_b  = (const float*)d_in[12];
  const float* n2_g  = (const float*)d_in[13];
  const float* n2_b  = (const float*)d_in[14];
  const float* n3_g  = (const float*)d_in[15];
  const float* n3_b  = (const float*)d_in[16];
  const float* ff_w1 = (const float*)d_in[17];
  const float* ff_b1 = (const float*)d_in[18];
  const float* ff_w2 = (const float*)d_in[19];
  const float* ff_b2 = (const float*)d_in[20];

  char* ws = (char*)d_ws;
  size_t off = 0;
  auto alloc = [&](size_t bytes) -> void* {
    void* p = ws + off; off += (bytes + 255) & ~(size_t)255; return p;
  };
  bf16_t* q_bf  = (bf16_t*)alloc((size_t)M_ * E_ * 2);
  bf16_t* p_bf  = (bf16_t*)alloc((size_t)B_ * P_ * E_ * 2);
  bf16_t* i_bf  = (bf16_t*)alloc((size_t)B_ * KL_ * E_ * 2);
  bf16_t* wt_tq = (bf16_t*)alloc((size_t)E_ * E_ * 2);
  bf16_t* wt_tk = (bf16_t*)alloc((size_t)E_ * E_ * 2);
  bf16_t* wt_tv = (bf16_t*)alloc((size_t)E_ * E_ * 2);
  bf16_t* wt_sq = (bf16_t*)alloc((size_t)E_ * E_ * 2);
  bf16_t* wt_sk = (bf16_t*)alloc((size_t)E_ * E_ * 2);
  bf16_t* wt_sv = (bf16_t*)alloc((size_t)E_ * E_ * 2);
  bf16_t* wt_so = (bf16_t*)alloc((size_t)E_ * E_ * 2);
  bf16_t* wt_f1 = (bf16_t*)alloc((size_t)FF_ * E_ * 2);
  bf16_t* wt_f2 = (bf16_t*)alloc((size_t)E_ * FF_ * 2);
  bf16_t* Qp    = (bf16_t*)alloc((size_t)M_ * E_ * 2);
  bf16_t* Kp    = (bf16_t*)alloc((size_t)B_ * P_ * E_ * 2);
  bf16_t* Vp    = (bf16_t*)alloc((size_t)B_ * P_ * E_ * 2);
  bf16_t* VTb   = (bf16_t*)alloc((size_t)B_ * H_ * D_ * NPAD_ * 2);
  float*  O1    = (float*) alloc((size_t)M_ * E_ * 4);
  float*  x1f   = (float*) alloc((size_t)M_ * E_ * 4);
  bf16_t* x1b   = (bf16_t*)alloc((size_t)M_ * E_ * 2);
  bf16_t* O2b   = (bf16_t*)alloc((size_t)M_ * E_ * 2);
  float*  C2    = (float*) alloc((size_t)M_ * E_ * 4);
  float*  x2f   = (float*) alloc((size_t)M_ * E_ * 4);
  bf16_t* x2b   = (bf16_t*)alloc((size_t)M_ * E_ * 2);
  bf16_t* ff1b  = (bf16_t*)alloc((size_t)M_ * FF_ * 2);
  float*  ff2   = (float*) alloc((size_t)M_ * E_ * 4);

  const int T = 256;
  // 1) inputs -> bf16
  k_f32_to_bf16<<<cdiv_i((long)M_ * E_, T), T, 0, stream>>>(query, q_bf, M_ * E_);
  k_f32_to_bf16<<<cdiv_i((long)B_ * P_ * E_, T), T, 0, stream>>>(prev, p_bf, B_ * P_ * E_);
  k_f32_to_bf16<<<cdiv_i((long)B_ * KL_ * E_, T), T, 0, stream>>>(img, i_bf, B_ * KL_ * E_);
  // 2) weight transposes -> bf16 (N x K)
  k_wtrans<<<cdiv_i((long)E_ * E_, T), T, 0, stream>>>(t_wq, wt_tq, E_, E_);
  k_wtrans<<<cdiv_i((long)E_ * E_, T), T, 0, stream>>>(t_wk, wt_tk, E_, E_);
  k_wtrans<<<cdiv_i((long)E_ * E_, T), T, 0, stream>>>(t_wv, wt_tv, E_, E_);
  k_wtrans<<<cdiv_i((long)E_ * E_, T), T, 0, stream>>>(s_wq, wt_sq, E_, E_);
  k_wtrans<<<cdiv_i((long)E_ * E_, T), T, 0, stream>>>(s_wk, wt_sk, E_, E_);
  k_wtrans<<<cdiv_i((long)E_ * E_, T), T, 0, stream>>>(s_wv, wt_sv, E_, E_);
  k_wtrans<<<cdiv_i((long)E_ * E_, T), T, 0, stream>>>(s_wo, wt_so, E_, E_);
  k_wtrans<<<cdiv_i((long)FF_ * E_, T), T, 0, stream>>>(ff_w1, wt_f1, E_, FF_);
  k_wtrans<<<cdiv_i((long)E_ * FF_, T), T, 0, stream>>>(ff_w2, wt_f2, FF_, E_);

  auto gemm_blocks = [&](int M, int N) {
    long waves = (long)cdiv_i(M, 16) * (N >> 6);
    return cdiv_i(waves, 8);
  };
  // 3) temporal Q/K/V projections
  k_gemm<<<gemm_blocks(M_, E_), T, 0, stream>>>(q_bf, wt_tq, nullptr, nullptr, Qp, M_, E_, E_, 0);
  k_gemm<<<gemm_blocks(B_ * P_, E_), T, 0, stream>>>(p_bf, wt_tk, nullptr, nullptr, Kp, B_ * P_, E_, E_, 0);
  k_gemm<<<gemm_blocks(B_ * P_, E_), T, 0, stream>>>(p_bf, wt_tv, nullptr, nullptr, Vp, B_ * P_, E_, E_, 0);
  k_vtrans<<<cdiv_i((long)B_ * H_ * D_ * NPAD_, T), T, 0, stream>>>(Vp, VTb, P_);
  // 4) temporal attention, scale = sqrt(E) = 16
  {
    int blocks = cdiv_i((long)B_ * H_ * cdiv_i(Q_, 32), 8);
    k_flash<<<blocks, T, 0, stream>>>(Qp, Kp, VTb, O1, nullptr, Q_, P_, 1.0f / 16.0f);
  }
  // 5) x1 = LN(attn + query)
  k_add_ln<<<cdiv_i(M_, 8), T, 0, stream>>>(O1, query, n1_g, n1_b, x1f, x1b, M_);
  // 6) spatial Q/K/V projections
  k_gemm<<<gemm_blocks(M_, E_), T, 0, stream>>>(x1b, wt_sq, nullptr, nullptr, Qp, M_, E_, E_, 0);
  k_gemm<<<gemm_blocks(B_ * KL_, E_), T, 0, stream>>>(i_bf, wt_sk, nullptr, nullptr, Kp, B_ * KL_, E_, E_, 0);
  k_gemm<<<gemm_blocks(B_ * KL_, E_), T, 0, stream>>>(i_bf, wt_sv, nullptr, nullptr, Vp, B_ * KL_, E_, E_, 0);
  k_vtrans<<<cdiv_i((long)B_ * H_ * D_ * NPAD_, T), T, 0, stream>>>(Vp, VTb, KL_);
  // 7) spatial attention, scale = sqrt(D) = sqrt(32)
  {
    int blocks = cdiv_i((long)B_ * H_ * cdiv_i(Q_, 32), 8);
    k_flash<<<blocks, T, 0, stream>>>(Qp, Kp, VTb, nullptr, O2b, Q_, KL_, 0.17677669529663687f);
  }
  // 8) output projection + bias
  k_gemm<<<gemm_blocks(M_, E_), T, 0, stream>>>(O2b, wt_so, s_bo, C2, nullptr, M_, E_, E_, 0);
  // 9) x2 = LN(cross + x1)
  k_add_ln<<<cdiv_i(M_, 8), T, 0, stream>>>(C2, x1f, n2_g, n2_b, x2f, x2b, M_);
  // 10) FFN
  k_gemm<<<gemm_blocks(M_, FF_), T, 0, stream>>>(x2b, wt_f1, ff_b1, nullptr, ff1b, M_, FF_, E_, 1);
  k_gemm<<<gemm_blocks(M_, E_), T, 0, stream>>>(ff1b, wt_f2, ff_b2, ff2, nullptr, M_, E_, FF_, 0);
  // 11) out = LN(ff + x2)
  k_add_ln<<<cdiv_i(M_, 8), T, 0, stream>>>(ff2, x2f, n3_g, n3_b, (float*)d_out, nullptr, M_);
}
